// variance_network_953482739897
// MI455X (gfx1250) — matrix operand order repaired and checked by reading, not compiled, not run on hardware
//
#include <hip/hip_runtime.h>
#include <hip/hip_bf16.h>

typedef __attribute__((ext_vector_type(16))) __bf16 v16bf;
typedef __attribute__((ext_vector_type(8)))  float  v8f;

#define M_TOTAL 32768
#define NPTS    2048
#define NF      512
#define KPAD    544          // 17 * 32, covers din=524 (stage1) and 513 (stage2/3)
#define LDT     40           // LDS tile row stride in elements (32 + 8 pad, 80B = 16B aligned)

#if __has_builtin(__builtin_amdgcn_s_wait_asynccnt)
#define WAIT_ASYNC(n) __builtin_amdgcn_s_wait_asynccnt(n)
#else
#define WAIT_ASYNC(n) asm volatile("s_wait_asynccnt %0" :: "i"(n) : "memory")
#endif

// Direct global -> LDS async copy (16B per lane), GVS addressing, tracked by ASYNCcnt.
__device__ __forceinline__ void async_g2l_b128(unsigned lds_byte_addr,
                                               const void* gbase,
                                               unsigned gbyte_off) {
    asm volatile("global_load_async_to_lds_b128 %0, %1, %2"
                 :: "v"(lds_byte_addr), "v"(gbyte_off),
                    "s"((unsigned long long)(size_t)gbase)
                 : "memory");
}

// ---------------- latent -> bf16 into X columns [0,512) ----------------
__global__ __launch_bounds__(256) void fill_latent_kernel(const float* __restrict__ latent,
                                                          __hip_bfloat16* __restrict__ X) {
    int idx = blockIdx.x * 256 + threadIdx.x;         // over 32768*512
    int row = idx >> 9;
    int col = idx & 511;
    X[row * KPAD + col] = __float2bfloat16(latent[idx]);
}

// ---------------- KNN (top-2 incl. self) + 3x3 outer product -> X cols [512,544) ----------------
__global__ __launch_bounds__(256) void knn_cov_kernel(const float* __restrict__ pc,
                                                      __hip_bfloat16* __restrict__ X) {
    const int lane = threadIdx.x & 31;
    const int wid  = threadIdx.x >> 5;
    const int p = blockIdx.x * 8 + wid;               // one wave per point, 4096 blocks
    const int b = p >> 11;
    const int i = p & (NPTS - 1);
    const float* base = pc + (size_t)b * NPTS * 3;
    const float xi0 = base[i*3+0], xi1 = base[i*3+1], xi2 = base[i*3+2];

    float d0 = 1e30f, d1 = 1e30f;
    int   i0 = 1 << 30, i1 = 1 << 30;
    for (int j = lane; j < NPTS; j += 32) {
        float dx = base[j*3+0] - xi0;
        float dy = base[j*3+1] - xi1;
        float dz = base[j*3+2] - xi2;
        float d = dx*dx + dy*dy + dz*dz;
        if ((d < d0) || (d == d0 && j < i0)) { d1 = d0; i1 = i0; d0 = d; i0 = j; }
        else if ((d < d1) || (d == d1 && j < i1)) { d1 = d; i1 = j; }
    }
    #pragma unroll
    for (int off = 16; off >= 1; off >>= 1) {
        float od0 = __shfl_xor(d0, off, 32);
        int   oi0 = __shfl_xor(i0, off, 32);
        float od1 = __shfl_xor(d1, off, 32);
        int   oi1 = __shfl_xor(i1, off, 32);
        if ((od0 < d0) || (od0 == d0 && oi0 < i0)) { d1 = d0; i1 = i0; d0 = od0; i0 = oi0; }
        else if ((od0 < d1) || (od0 == d1 && oi0 < i1)) { d1 = od0; i1 = oi0; }
        if ((od1 < d0) || (od1 == d0 && oi1 < i0)) { d1 = d0; i1 = i0; d0 = od1; i0 = oi1; }
        else if ((od1 < d1) || (od1 == d1 && oi1 < i1)) { d1 = od1; i1 = oi1; }
    }
    if (lane == 0) {
        const float* a = base + i0 * 3;   // top-1 (normally self)
        const float* c = base + i1 * 3;   // top-2 (nearest other)
        __hip_bfloat16* xrow = X + (size_t)p * KPAD + 512;
        xrow[0] = __float2bfloat16(xi0);
        xrow[1] = __float2bfloat16(xi1);
        xrow[2] = __float2bfloat16(xi2);
        #pragma unroll
        for (int r = 0; r < 3; ++r)
            #pragma unroll
            for (int cc = 0; cc < 3; ++cc)
                xrow[3 + r*3 + cc] = __float2bfloat16(a[r] * c[cc]);
        #pragma unroll
        for (int k = 12; k < KPAD - 512; ++k) xrow[k] = __float2bfloat16(0.0f);
    }
}

// ---------------- stage 2/3: out scalar -> X col 512, zero the rest ----------------
__global__ __launch_bounds__(256) void prep_extras_kernel(const float* __restrict__ out,
                                                          __hip_bfloat16* __restrict__ X) {
    int p = blockIdx.x * 256 + threadIdx.x;           // 32768 points
    __hip_bfloat16* xrow = X + (size_t)p * KPAD + 512;
    xrow[0] = __float2bfloat16(out[p]);
    #pragma unroll
    for (int k = 1; k < KPAD - 512; ++k) xrow[k] = __float2bfloat16(0.0f);
}

// ---------------- weight prep: f32 (din x 512) -> bf16 transposed (512 x kpad), zero-padded ----
__global__ __launch_bounds__(256) void wprep_kernel(const float* __restrict__ W,
                                                    __hip_bfloat16* __restrict__ WT,
                                                    int din, int kpad) {
    int idx = blockIdx.x * 256 + threadIdx.x;         // over 512*kpad
    int n = idx / kpad;
    int k = idx - n * kpad;
    float v = (k < din) ? W[(size_t)k * NF + n] : 0.0f;
    WT[idx] = __float2bfloat16(v);
}

// ---------------- WMMA GEMM: out = relu(A(MxK) @ BT(NxK)^T + bias [+ res]) , bf16 out -------
// Double-buffered LDS fed by GLOBAL_LOAD_ASYNC_TO_LDS_B128 (ASYNCcnt pipeline).
__global__ __launch_bounds__(256) void gemm_wmma_kernel(
    const __hip_bfloat16* __restrict__ A,   // M x K row-major bf16
    const __hip_bfloat16* __restrict__ BT,  // 512 x K row-major bf16 (transposed weight)
    const float* __restrict__ bias,         // 512
    const __hip_bfloat16* __restrict__ res, // M x 512 bf16 or nullptr
    __hip_bfloat16* __restrict__ out,       // M x 512 bf16
    int K)
{
    __shared__ unsigned short As[2][128 * LDT];
    __shared__ unsigned short Bs[2][128 * LDT];

    const int tid  = threadIdx.x;
    const int lane = tid & 31;
    const int wid  = tid >> 5;              // 8 wave32s
    const int wm   = wid & 3;               // 4 waves along M
    const int wn   = wid >> 2;              // 2 waves along N
    const int m0   = blockIdx.x * 128;
    const int n0   = blockIdx.y * 128;
    const int nk   = K >> 5;

    // per-thread copy coords: 512 16B-chunks per 128x32 tile, 2 per thread
    const int r0 = tid >> 2;                // chunk 0 row
    const int c0 = (tid & 3) * 8;           // chunk 0 col (elements)
    const int r1 = (tid + 256) >> 2;        // chunk 1 row
    const int c1 = ((tid + 256) & 3) * 8;

    // issue async copies for K-chunk kc into LDS buffer `buf` (4 per lane)
    auto issue_chunk = [&](int kc, int buf) {
        const int kbase = kc * 32;
        async_g2l_b128((unsigned)(size_t)&As[buf][r0 * LDT + c0], A,
                       (unsigned)(((m0 + r0) * K + kbase + c0) * 2));
        async_g2l_b128((unsigned)(size_t)&Bs[buf][r0 * LDT + c0], BT,
                       (unsigned)(((n0 + r0) * K + kbase + c0) * 2));
        async_g2l_b128((unsigned)(size_t)&As[buf][r1 * LDT + c1], A,
                       (unsigned)(((m0 + r1) * K + kbase + c1) * 2));
        async_g2l_b128((unsigned)(size_t)&Bs[buf][r1 * LDT + c1], BT,
                       (unsigned)(((n0 + r1) * K + kbase + c1) * 2));
    };

    v8f acc[2][4] = {};

    issue_chunk(0, 0);
    for (int k = 0; k < nk; ++k) {
        // prefetch next chunk into the other buffer (last iter: redundant copy into
        // the idle buffer so the wait immediate stays constant — never read again)
        const int kn = (k + 1 < nk) ? (k + 1) : k;
        issue_chunk(kn, (k + 1) & 1);
        // newest 4 in flight are chunk k+1; async loads complete in order,
        // so <=4 guarantees chunk k has fully landed in LDS
        WAIT_ASYNC(4);
        __syncthreads();

        const unsigned short* cA = As[k & 1];
        const unsigned short* cB = Bs[k & 1];

        // A fragments: lane<16 -> row m, K {0..7,16..23}; lane>=16 -> K {8..15,24..31}
        const int arow = wm * 32 + (lane & 15);
        const int akb  = (lane >> 4) * 8;
        union { v16bf v; uint4 u[2]; } af[2];
        #pragma unroll
        for (int mi = 0; mi < 2; ++mi) {
            int r = arow + mi * 16;
            af[mi].u[0] = *(const uint4*)&cA[r * LDT + akb];
            af[mi].u[1] = *(const uint4*)&cA[r * LDT + akb + 16];
        }
        // B fragments: lane<16 -> col n, K {0..7,16..23}; lane>=16 -> K {8..15,24..31}
        const int bcol = wn * 64 + (lane & 15);
        const int bkb  = (lane >> 4) * 8;
        union { v16bf v; uint4 u[2]; } bf[4];
        #pragma unroll
        for (int ni = 0; ni < 4; ++ni) {
            int r = bcol + ni * 16;
            bf[ni].u[0] = *(const uint4*)&cB[r * LDT + bkb];
            bf[ni].u[1] = *(const uint4*)&cB[r * LDT + bkb + 16];
        }

        #pragma unroll
        for (int mi = 0; mi < 2; ++mi)
            #pragma unroll
            for (int ni = 0; ni < 4; ++ni)
                acc[mi][ni] = __builtin_amdgcn_wmma_f32_16x16x32_bf16(
                    false, af[mi].v, false, bf[ni].v,
                    (short)0, acc[mi][ni], false, false);
        __syncthreads();   // frag reads done (ds waits precede wmma) -> buffer reusable
    }

    // epilogue: bias + optional residual + ReLU, store bf16
    #pragma unroll
    for (int mi = 0; mi < 2; ++mi) {
        const int rowbase = m0 + wm * 32 + mi * 16 + ((lane >> 4) << 3);
        #pragma unroll
        for (int ni = 0; ni < 4; ++ni) {
            const int col = n0 + wn * 64 + ni * 16 + (lane & 15);
            const float bv = bias[col];
            #pragma unroll
            for (int j = 0; j < 8; ++j) {
                const int row = rowbase + j;
                float c = acc[mi][ni][j] + bv;
                if (res) c += __bfloat162float(res[(size_t)row * NF + col]);
                c = fmaxf(c, 0.0f);
                out[(size_t)row * NF + col] = __float2bfloat16(c);
            }
        }
    }
}

// ---------------- skinny 512 -> 1: out[row] (+)= H2[row,:] . wo + bo ----------------
__global__ __launch_bounds__(256) void rowdot_kernel(const __hip_bfloat16* __restrict__ H,
                                                     const float* __restrict__ wo,
                                                     const float* __restrict__ bo,
                                                     float* __restrict__ out,
                                                     int accumulate) {
    const int lane = threadIdx.x & 31;
    const int wid  = threadIdx.x >> 5;
    const int row  = blockIdx.x * 8 + wid;
    const __hip_bfloat16* hrow = H + (size_t)row * NF;
    float s = 0.0f;
    #pragma unroll
    for (int k = lane; k < NF; k += 32)
        s += __bfloat162float(hrow[k]) * wo[k];
    #pragma unroll
    for (int off = 16; off >= 1; off >>= 1)
        s += __shfl_xor(s, off, 32);
    if (lane == 0) {
        float v = s + bo[0];
        out[row] = accumulate ? (out[row] + v) : v;
    }
}

extern "C" void kernel_launch(void* const* d_in, const int* in_sizes, int n_in,
                              void* d_out, int out_size, void* d_ws, size_t ws_size,
                              hipStream_t stream) {
    const float* latent = (const float*)d_in[0];
    const float* pc     = (const float*)d_in[1];
    float* out = (float*)d_out;                       // 32768 f32

    // scratch layout
    char* ws = (char*)d_ws;
    auto take = [&](size_t bytes) { void* p = (void*)ws; ws += (bytes + 255) & ~(size_t)255; return p; };
    __hip_bfloat16* X  = (__hip_bfloat16*)take((size_t)M_TOTAL * KPAD * 2);
    __hip_bfloat16* H  = (__hip_bfloat16*)take((size_t)M_TOTAL * NF * 2);
    __hip_bfloat16* H2 = (__hip_bfloat16*)take((size_t)M_TOTAL * NF * 2);
    __hip_bfloat16* WaT[3];
    __hip_bfloat16* WrT[3];
    for (int s = 0; s < 3; ++s) {
        WaT[s] = (__hip_bfloat16*)take((size_t)NF * KPAD * 2);
        WrT[s] = (__hip_bfloat16*)take((size_t)NF * NF * 2);
    }

    // build bf16 input matrix: latent + knn/cov features
    fill_latent_kernel<<<(M_TOTAL * NF) / 256, 256, 0, stream>>>(latent, X);
    knn_cov_kernel<<<M_TOTAL / 8, 256, 0, stream>>>(pc, X);

    // weight prep (convert + transpose + zero-pad)
    for (int s = 0; s < 3; ++s) {
        const float* wa = (const float*)d_in[2 + 6 * s];
        const float* wr = (const float*)d_in[4 + 6 * s];
        int din_a = (s == 0) ? (NF + 12) : (NF + 1);
        wprep_kernel<<<(NF * KPAD) / 256, 256, 0, stream>>>(wa, WaT[s], din_a, KPAD);
        wprep_kernel<<<(NF * NF) / 256, 256, 0, stream>>>(wr, WrT[s], NF, NF);
    }

    // three residual MLP stages
    for (int s = 0; s < 3; ++s) {
        const float* ba = (const float*)d_in[3 + 6 * s];
        const float* br = (const float*)d_in[5 + 6 * s];
        const float* wo = (const float*)d_in[6 + 6 * s];
        const float* bo = (const float*)d_in[7 + 6 * s];
        if (s > 0)
            prep_extras_kernel<<<M_TOTAL / 256, 256, 0, stream>>>(out, X);
        // H = relu(X @ Wa + ba)
        gemm_wmma_kernel<<<dim3(M_TOTAL / 128, NF / 128), 256, 0, stream>>>(
            X, WaT[s], ba, nullptr, H, KPAD);
        // H2 = relu(H + H @ Wr + br)
        gemm_wmma_kernel<<<dim3(M_TOTAL / 128, NF / 128), 256, 0, stream>>>(
            H, WrT[s], br, H, H2, NF);
        // out (+)= H2 @ wo + bo
        rowdot_kernel<<<M_TOTAL / 8, 256, 0, stream>>>(H2, wo, bo, out, s > 0);
    }
}
